// QLSTMGen303_65481071407804
// MI455X (gfx1250) — matrix operand order repaired and checked by reading, not compiled, not run on hardware
//
#include <hip/hip_runtime.h>
#include <hip/hip_bf16.h>
#include <stdint.h>

// ---------------- problem dims ----------------
#define T_STEPS 512
#define BATCH   64
#define IN_DIM  512
#define HID     512
#define NCOLS   2048            // 4 gates * HID
#define NWG     16              // recurrence workgroups (1 per WGP)
#define JPW     (HID / NWG)     // 32 hidden units per workgroup

typedef __bf16 bf16_t;
typedef __bf16 v16bf __attribute__((ext_vector_type(16)));
typedef __bf16 v8bf  __attribute__((ext_vector_type(8)));
typedef float  v8f   __attribute__((ext_vector_type(8)));
typedef unsigned int u32x4 __attribute__((ext_vector_type(4)));
typedef int          i32x4 __attribute__((ext_vector_type(4)));
typedef int          i32x8 __attribute__((ext_vector_type(8)));

static __device__ __forceinline__ v8f wmma_bf16(v16bf a, v16bf b, v8f c) {
  // D(f32 16x16) = A(bf16 16x32) * B(bf16 32x16) + C
  return __builtin_amdgcn_wmma_f32_16x16x32_bf16(false, a, false, b, (short)0, c,
                                                 false, false);
}

// A-fragment (16x32 bf16, contiguous K): lanes 0-15 hold K = kb+{0..15},
// lanes 16-31 hold K = kb+{16..31} for row M = m0 + lane%16.
static __device__ __forceinline__ v16bf make_a_frag(const bf16_t* rowp, int kb, int halfA) {
  v8bf lo = *(const v8bf*)(rowp + kb + 2 * halfA);
  v8bf hi = *(const v8bf*)(rowp + kb + 2 * halfA + 8);
  v16bf a;
#pragma unroll
  for (int e = 0; e < 8; ++e) { a[e] = lo[e]; a[e + 8] = hi[e]; }
  return a;
}

// fast sigmoid / tanh on the serial path: 1 v_exp + 1 v_rcp each
static __device__ __forceinline__ float sigmoidf_fast(float x) {
  return __frcp_rn(1.0f + __expf(-x));
}
static __device__ __forceinline__ float tanhf_fast(float x) {
  float e = __expf(2.0f * x);
  return (e - 1.0f) * __frcp_rn(e + 1.0f);
}

// ------------------------------------------------------------------
// Prep: split+transpose weights to bf16 W^T layouts, combine biases,
// zero h double-buffer and the step-barrier counter (deterministic).
// ------------------------------------------------------------------
__global__ void k_prep_weights(const float* Wf, const float* bf_,
                               const float* Wi, const float* bi_,
                               const float* Wg, const float* bg_,
                               const float* Wo, const float* bo_,
                               bf16_t* __restrict__ WxT, bf16_t* __restrict__ WhT,
                               float* __restrict__ bias, bf16_t* __restrict__ hbuf,
                               unsigned* __restrict__ bar) {
  const size_t total = (size_t)NCOLS * HID;
  const size_t stride = (size_t)gridDim.x * blockDim.x;
  for (size_t idx = (size_t)blockIdx.x * blockDim.x + threadIdx.x; idx < total; idx += stride) {
    int n = (int)(idx >> 9);        // output column 0..2047
    int k = (int)(idx & (HID - 1)); // reduction index 0..511
    int gate = n >> 9;              // 0=f 1=i 2=g 3=o
    int j = n & (HID - 1);
    const float* W; const float* bsrc;
    if (gate == 0)      { W = Wf; bsrc = bf_; }
    else if (gate == 1) { W = Wi; bsrc = bi_; }
    else if (gate == 2) { W = Wg; bsrc = bg_; }
    else                { W = Wo; bsrc = bo_; }
    // W is [cin=1024, HID] row-major; rows [0,512) = x part, [512,1024) = h part
    WxT[(size_t)n * HID + k] = (bf16_t)W[(size_t)k * HID + j];
    WhT[(size_t)n * HID + k] = (bf16_t)W[(size_t)(k + IN_DIM) * HID + j];
    if (k == 0) bias[n] = bsrc[j];
  }
  for (size_t idx = (size_t)blockIdx.x * blockDim.x + threadIdx.x;
       idx < (size_t)2 * BATCH * HID; idx += stride)
    hbuf[idx] = (bf16_t)0.0f;
  if (blockIdx.x == 0 && threadIdx.x == 0) *bar = 0u;
}

__global__ void k_cvt_inputs(const float* __restrict__ x, bf16_t* __restrict__ xb, size_t n) {
  const size_t stride = (size_t)gridDim.x * blockDim.x;
  for (size_t i = (size_t)blockIdx.x * blockDim.x + threadIdx.x; i < n; i += stride)
    xb[i] = (bf16_t)x[i];
}

// ------------------------------------------------------------------
// Phase 1: XW[tb, n] = Xbf[tb, :] @ WxT[n, :]   (bf16 WMMA, f32 acc)
// One wave computes a 16(M) x 64(N) strip; the A fragment is register
// double-buffered so global loads overlap the 4 WMMAs per k-step.
// ------------------------------------------------------------------
__global__ __launch_bounds__(256) void k_xgemm(const bf16_t* __restrict__ Xbf,
                                               const bf16_t* __restrict__ WxT,
                                               float* __restrict__ XW) {
  const int lane = threadIdx.x & 31;
  const int wid  = (int)((blockIdx.x * blockDim.x + threadIdx.x) >> 5);
  const int n_groups = NCOLS / 64;         // 32
  const int m_tile = wid / n_groups;       // 0..2047
  const int n_group = wid % n_groups;
  if (m_tile >= (T_STEPS * BATCH) / 16) return;

  const int m0 = m_tile * 16;
  const int n0 = n_group * 64;
  const int halfA = (lane >> 4) << 3;      // 0 or 8
  const int halfB = (lane >> 4) << 4;      // 0 or 16
  const int rowA  = m0 + (lane & 15);

  v8f acc[4];
#pragma unroll
  for (int nt = 0; nt < 4; ++nt)
#pragma unroll
    for (int r = 0; r < 8; ++r) acc[nt][r] = 0.0f;

  const bf16_t* arow = Xbf + (size_t)rowA * IN_DIM;
  v16bf a_cur = make_a_frag(arow, 0, halfA);
#pragma unroll
  for (int kb = 0; kb < IN_DIM; kb += 32) {
    v16bf a_next;
    if (kb + 32 < IN_DIM) a_next = make_a_frag(arow, kb + 32, halfA);
#pragma unroll
    for (int nt = 0; nt < 4; ++nt) {
      int n = n0 + nt * 16 + (lane & 15);
      v16bf b = *(const v16bf*)(WxT + (size_t)n * IN_DIM + kb + halfB);
      acc[nt] = wmma_bf16(a_cur, b, acc[nt]);
    }
    a_cur = a_next;
  }
  const int rowC = m0 + 8 * (lane >> 4);
#pragma unroll
  for (int nt = 0; nt < 4; ++nt) {
    int col = n0 + nt * 16 + (lane & 15);
#pragma unroll
    for (int r = 0; r < 8; ++r)
      XW[(size_t)(rowC + r) * NCOLS + col] = acc[nt][r];
  }
}

// ------------------------------------------------------------------
// Phase 2: persistent recurrence. 16 WGs; each owns 32 hidden units
// (all 4 gates). Wh^T slice staged once into LDS via TDM; c-state in
// VGPRs; h exchanged through a double-buffered L2-resident array with
// a per-step global barrier (+ cluster barrier on the CDNA5 path).
// ------------------------------------------------------------------
__global__ __launch_bounds__(256) void k_recur(const float* __restrict__ XW,
                                               const bf16_t* __restrict__ WhT,
                                               const float* __restrict__ bias,
                                               bf16_t* __restrict__ hbuf,
                                               float* __restrict__ out,
                                               unsigned* __restrict__ bar) {
  extern __shared__ bf16_t smem[];          // [4*JPW][HID] bf16 = 128 KB
  const int wg   = blockIdx.x;              // 0..15
  const int lane = threadIdx.x & 31;
  const int wid  = threadIdx.x >> 5;        // 0..7
  const int m_tile = wid & 3;               // batch tile
  const int j_tile = wid >> 2;              // hidden sub-tile (0..1)
  const int js = wg * JPW;

  // ---- stage Wh^T slice into LDS: 4 tiles of 32 rows x 512 bf16 (TDM) ----
#if __has_builtin(__builtin_amdgcn_tensor_load_to_lds) && __has_builtin(__builtin_amdgcn_s_wait_tensorcnt)
  if (wid == 0) {
#pragma unroll
    for (int g = 0; g < 4; ++g) {
      unsigned long long ga =
          (unsigned long long)(uintptr_t)(WhT + (size_t)(g * HID + js) * HID);
      unsigned lds_off = (unsigned)(g * JPW * HID * sizeof(bf16_t));
      u32x4 g0;
      g0[0] = 1u;                                       // count=1, user descriptor
      g0[1] = lds_off;                                  // lds_addr (bytes)
      g0[2] = (unsigned)ga;                             // global_addr lo
      g0[3] = (unsigned)((ga >> 32) & 0x01FFFFFFull) | (2u << 30); // hi + type=2
      i32x8 g1;
      g1[0] = (int)(1u << 16);          // data_size=1 (2B), wg_mask=0, no pad
      g1[1] = (int)((unsigned)HID << 16);   // tensor_dim0[15:0]=512 @bit48
      g1[2] = (int)((unsigned)NCOLS << 16); // dim0 hi=0 | tensor_dim1[15:0]=2048
      g1[3] = (int)((unsigned)HID << 16);   // dim1 hi=0 | tile_dim0=512
      g1[4] = (int)JPW;                     // tile_dim1=32, tile_dim2=0
      g1[5] = (int)HID;                     // tensor_dim0_stride=512 (lo32)
      g1[6] = 0;                            // stride hi, dim1_stride lo
      g1[7] = 0;
      i32x4 z4 = {0, 0, 0, 0};
#if defined(__clang_major__) && (__clang_major__ >= 23)
      i32x8 z8 = {0, 0, 0, 0, 0, 0, 0, 0};
      __builtin_amdgcn_tensor_load_to_lds(g0, g1, z4, z4, z8, 0);
#else
      __builtin_amdgcn_tensor_load_to_lds(g0, g1, z4, z4, 0);
#endif
    }
    __builtin_amdgcn_s_wait_tensorcnt(0);
    asm volatile("" ::: "memory");
  }
#else
  for (int i = threadIdx.x; i < 4 * JPW * HID; i += blockDim.x) {
    int g = i / (JPW * HID);
    int rem = i % (JPW * HID);
    smem[i] = WhT[(size_t)(g * HID + js + rem / HID) * HID + (rem % HID)];
  }
#endif
  __syncthreads();

  const int j_local = j_tile * 16 + (lane & 15);  // 0..31 within WG slice
  const int jg = js + j_local;                    // global hidden index
  const int rowA = m_tile * 16 + (lane & 15);     // A-matrix row (batch)
  const int rowC = m_tile * 16 + 8 * (lane >> 4); // C-matrix row base
  const int halfA = (lane >> 4) << 3;
  const int halfB = (lane >> 4) << 4;

  float biasv[4];
#pragma unroll
  for (int g = 0; g < 4; ++g) biasv[g] = bias[g * HID + jg];

  v8f cfrag, hlast;
#pragma unroll
  for (int r = 0; r < 8; ++r) { cfrag[r] = 0.0f; hlast[r] = 0.0f; }

  for (int t = 0; t < T_STEPS; ++t) {
    const bf16_t* hread  = hbuf + (size_t)(t & 1) * BATCH * HID;
    bf16_t*       hwrite = hbuf + (size_t)((t + 1) & 1) * BATCH * HID;
    const float*  xwbase = XW + (size_t)t * BATCH * NCOLS;
    float*        obase  = out + (size_t)t * BATCH * HID;

    // Issue the (independent) XW gate loads first so their latency is
    // hidden under the whole recurrent GEMM. Single-use stream -> NT.
    float xwv[4][8];
#pragma unroll
    for (int r = 0; r < 8; ++r) {
      const float* xwr = xwbase + (size_t)(rowC + r) * NCOLS + jg;
#pragma unroll
      for (int g = 0; g < 4; ++g)
        xwv[g][r] = __builtin_nontemporal_load(xwr + g * HID);
    }
    if (t + 1 < T_STEPS)
      __builtin_prefetch(xwbase + (size_t)BATCH * NCOLS + (size_t)rowC * NCOLS + jg, 0, 1);

    v8f acc[4];
#pragma unroll
    for (int g = 0; g < 4; ++g)
#pragma unroll
      for (int r = 0; r < 8; ++r) acc[g][r] = 0.0f;

    // K loop: A frag register-double-buffered; B frags from LDS.
    const bf16_t* arow = hread + (size_t)rowA * HID;
    v16bf a_cur = make_a_frag(arow, 0, halfA);
#pragma unroll
    for (int kb = 0; kb < HID; kb += 32) {
      v16bf a_next;
      if (kb + 32 < HID) a_next = make_a_frag(arow, kb + 32, halfA);
#pragma unroll
      for (int g = 0; g < 4; ++g) {
        const bf16_t* bp = smem + (size_t)(g * JPW + j_local) * HID + kb + halfB;
        v16bf b = *(const v16bf*)bp;                 // ds_load_b128 x2
        acc[g] = wmma_bf16(a_cur, b, acc[g]);
      }
      a_cur = a_next;
    }

#pragma unroll
    for (int r = 0; r < 8; ++r) {
      int b = rowC + r;
      float fx = acc[0][r] + xwv[0][r] + biasv[0];
      float ix = acc[1][r] + xwv[1][r] + biasv[1];
      float gx = acc[2][r] + xwv[2][r] + biasv[2];
      float ox = acc[3][r] + xwv[3][r] + biasv[3];
      float fg = sigmoidf_fast(fx);
      float ig = sigmoidf_fast(ix);
      float gg = tanhf_fast(gx);
      float og = sigmoidf_fast(ox);
      float c  = fg * cfrag[r] + ig * gg;
      cfrag[r] = c;
      float h  = og * tanhf_fast(c);
      hlast[r] = h;
      obase[(size_t)b * HID + jg]  = h;              // stacked output (fp32)
      hwrite[(size_t)b * HID + jg] = (bf16_t)h;      // next-step A matrix
    }

    // ---- cross-workgroup step barrier ----
    __threadfence();
    __syncthreads();
    if (threadIdx.x == 0) {
      atomicAdd(bar, 1u);
      unsigned tgt = (unsigned)(t + 1) * NWG;
      while (atomicAdd(bar, 0u) < tgt) __builtin_amdgcn_s_sleep(2);
    }
    __builtin_amdgcn_s_cluster_barrier();  // NOP when not cluster-launched
    __syncthreads();
    __threadfence();
  }

  // final (hx, cx) tail of d_out
  float* hx = out + (size_t)T_STEPS * BATCH * HID;
  float* cx = hx + (size_t)BATCH * HID;
#pragma unroll
  for (int r = 0; r < 8; ++r) {
    int b = rowC + r;
    hx[(size_t)b * HID + jg] = hlast[r];
    cx[(size_t)b * HID + jg] = cfrag[r];
  }
}

// ------------------------------------------------------------------
extern "C" void kernel_launch(void* const* d_in, const int* in_sizes, int n_in,
                              void* d_out, int out_size, void* d_ws, size_t ws_size,
                              hipStream_t stream) {
  (void)in_sizes; (void)n_in; (void)out_size; (void)ws_size;
  const float* x   = (const float*)d_in[0];
  const float* Wf  = (const float*)d_in[1];
  const float* bf_ = (const float*)d_in[2];
  const float* Wi  = (const float*)d_in[3];
  const float* bi_ = (const float*)d_in[4];
  const float* Wg  = (const float*)d_in[5];
  const float* bg_ = (const float*)d_in[6];
  const float* Wo  = (const float*)d_in[7];
  const float* bo_ = (const float*)d_in[8];

  char* ws = (char*)d_ws;
  size_t off = 0;
  auto take = [&](size_t bytes) { void* p = ws + off; off += (bytes + 255) & ~(size_t)255; return p; };
  bf16_t*  Xbf  = (bf16_t*)take((size_t)T_STEPS * BATCH * IN_DIM * sizeof(bf16_t)); // 32 MB
  bf16_t*  WxT  = (bf16_t*)take((size_t)NCOLS * IN_DIM * sizeof(bf16_t));           // 2 MB
  bf16_t*  WhT  = (bf16_t*)take((size_t)NCOLS * HID * sizeof(bf16_t));              // 2 MB
  float*   bias = (float*) take((size_t)NCOLS * sizeof(float));
  bf16_t*  hbuf = (bf16_t*)take((size_t)2 * BATCH * HID * sizeof(bf16_t));          // 128 KB
  unsigned* bar = (unsigned*)take(256);
  float*   XW   = (float*) take((size_t)T_STEPS * BATCH * NCOLS * sizeof(float));   // 256 MB

  k_prep_weights<<<1024, 256, 0, stream>>>(Wf, bf_, Wi, bi_, Wg, bg_, Wo, bo_,
                                           WxT, WhT, bias, hbuf, bar);
  k_cvt_inputs<<<8192, 256, 0, stream>>>(x, Xbf, (size_t)T_STEPS * BATCH * IN_DIM);
  k_xgemm<<<8192, 256, 0, stream>>>(Xbf, WxT, XW);
  k_recur<<<NWG, 256, 4 * JPW * HID * sizeof(bf16_t), stream>>>(
      XW, WhT, bias, hbuf, (float*)d_out, bar);
}
